// VMamba_10788957847756
// MI455X (gfx1250) — compile-verified, matrix operand order checked by reading, not run on hardware
//
#include <hip/hip_runtime.h>
#include <hip/hip_bf16.h>
#include <math.h>

// ---------------------------------------------------------------------------
// VMamba (SS2D) forward for MI455X / gfx1250, wave32, WMMA bf16->f32.
//   B=4, C_IN=64, D=128, D4=512, N=16, DR=4, H=W=64, L=4096
// Pipeline:
//   k_inproj      : WMMA GEMM [16384 x 64] x [64 x 256] -> xz
//   k_conv_u      : depthwise 3x3 + SiLU, scatter into 4 scan orders -> u
//   k_xdbl        : WMMA GEMM [16384 x 512] x [512 x 48(pad36)] -> xdbl
//                   weights staged to LDS once per block via TDM
//                   (tensor_load_to_lds + s_wait_tensorcnt)
//   k_scan_part   : chunked scan pass A: per-chunk decay product P and
//                   zero-init endpoint E            (2M threads, 64 steps)
//   k_scan_combine: sequential prefix over 64 chunks (32K threads)
//   k_scan_apply  : pass B from corrected start state, y via shfl-reduce,
//                   y + u*D_param written in-place over u
//   k_out         : merge 4 dirs + LayerNorm + SiLU(z) gate + WMMA out_proj
// ---------------------------------------------------------------------------

typedef __attribute__((ext_vector_type(16))) __bf16 v16bf;
typedef __attribute__((ext_vector_type(8)))  float  v8f;
typedef unsigned int __attribute__((ext_vector_type(4))) v4u;
typedef int          __attribute__((ext_vector_type(8))) v8i;
typedef int          __attribute__((ext_vector_type(4))) v4i;

#define BSZ   4
#define CIN   64
#define DD    128
#define D4    512
#define NST   16
#define HH    64
#define WW    64
#define LL    4096          // H*W
#define MROWS (BSZ*LL)      // 16384
#define E2D   256           // 2*D
#define XDW   48            // padded x_dbl width (36 -> 48)
#define NCH   64            // scan chunks
#define CHS   64            // steps per chunk (NCH*CHS == LL)

// K index inside a v16bf fragment for the 16-bit 16x16x32 WMMA layout:
// lanes 0-15 hold K-half 0, lanes 16-31 hold K-half 1.
__device__ __forceinline__ int kmap(int e, int half) {
    return (half << 3) + (e & 7) + ((e >> 3) << 4);
}

__device__ __forceinline__ float silu_f(float v) {
    return v / (1.0f + __expf(-v));
}

__device__ __forceinline__ float softplus_f(float v) {
    return (v > 20.0f) ? v : log1pf(__expf(v));
}

// ---------------------------------------------------------------------------
// Kernel 1: xz[p, e] = sum_c x[b, c, hw] * in_proj_w[e, c] + in_proj_b[e]
// One wave per 16(M) x 64(N) tile.  K = 64 (2 WMMA k-steps).
// ---------------------------------------------------------------------------
__global__ __launch_bounds__(32)
void k_inproj(const float* __restrict__ x, const float* __restrict__ w,
              const float* __restrict__ bias, float* __restrict__ xz)
{
    const int lane = threadIdx.x & 31;
    const int half = lane >> 4;
    const int mr   = lane & 15;
    const int p0   = blockIdx.x * 16;
    const int n0   = blockIdx.y * 64;

    const int p  = p0 + mr;
    const int b  = p >> 12;
    const int hw = p & (LL - 1);
    const float* xb = x + (size_t)b * CIN * LL + hw;   // x[(b*64+k)*4096 + hw]

    v8f acc[4] = {};
#pragma unroll
    for (int k0 = 0; k0 < CIN; k0 += 32) {
        v16bf a;
#pragma unroll
        for (int e = 0; e < 16; ++e)
            a[e] = (__bf16)xb[(size_t)(k0 + kmap(e, half)) * LL];
#pragma unroll
        for (int nt = 0; nt < 4; ++nt) {
            const int n = n0 + nt * 16 + mr;
            v16bf bf;
#pragma unroll
            for (int e = 0; e < 16; ++e)
                bf[e] = (__bf16)w[n * CIN + k0 + kmap(e, half)];
            acc[nt] = __builtin_amdgcn_wmma_f32_16x16x32_bf16(
                false, a, false, bf, (short)0, acc[nt], false, false);
        }
    }
#pragma unroll
    for (int nt = 0; nt < 4; ++nt) {
        const int n  = n0 + nt * 16 + mr;
        const float bv = bias[n];
#pragma unroll
        for (int r = 0; r < 8; ++r) {
            const int m = r + 8 * half;
            xz[(size_t)(p0 + m) * E2D + n] = acc[nt][r] + bv;
        }
    }
}

// ---------------------------------------------------------------------------
// Kernel 2: depthwise 3x3 conv (SAME) + bias + SiLU on xs = xz[:, :128],
// scatter result into the 4 direction-permuted slots of u (involutions).
// ---------------------------------------------------------------------------
__global__ void k_conv_u(const float* __restrict__ xz, const float* __restrict__ cw,
                         const float* __restrict__ cb, float* __restrict__ u)
{
    const int idx = blockIdx.x * blockDim.x + threadIdx.x;
    if (idx >= BSZ * DD * LL) return;
    const int l = idx & (LL - 1);
    const int d = (idx >> 12) & (DD - 1);
    const int b = idx >> 19;
    const int h = l >> 6, w = l & 63;

    float acc = cb[d];
#pragma unroll
    for (int i = 0; i < 3; ++i) {
        const int hh = h + i - 1;
        if ((unsigned)hh >= HH) continue;
#pragma unroll
        for (int j = 0; j < 3; ++j) {
            const int ww = w + j - 1;
            if ((unsigned)ww >= WW) continue;
            acc += xz[(size_t)((b * HH + hh) * WW + ww) * E2D + d] * cw[d * 9 + i * 3 + j];
        }
    }
    const float v  = silu_f(acc);
    const int   tl = ((l & 63) << 6) | (l >> 6);      // transpose(l)
    float* ub = u + (size_t)b * LL * D4;
    ub[(size_t)l           * D4 +          d] = v;
    ub[(size_t)tl          * D4 + DD     + d] = v;
    ub[(size_t)(4095 - l)  * D4 + 2 * DD + d] = v;
    ub[(size_t)(4095 - tl) * D4 + 3 * DD + d] = v;
}

// ---------------------------------------------------------------------------
// Kernel 3: xdbl[p, j] = sum_k u[p,k] * x_proj_w[j,k],  j < 36 (padded to 48).
// 8 waves / block; x_proj_w (36 x 512 f32 = 72 KB) staged into LDS once per
// block by the Tensor Data Mover, then B-fragments are gathered from LDS.
// ---------------------------------------------------------------------------
__global__ __launch_bounds__(256)
void k_xdbl(const float* __restrict__ u, const float* __restrict__ xpw,
            float* __restrict__ xdbl)
{
    __shared__ float wlds[36 * D4];            // 73,728 bytes

    const int lane = threadIdx.x & 31;
    const int wv   = threadIdx.x >> 5;         // wave id 0..7

    if (wv == 0) {
        // ---- TDM: DMA the whole 36x512 f32 weight tensor into LDS ----
        const unsigned long long ga = (unsigned long long)(uintptr_t)xpw;
        const unsigned int       la = (unsigned int)(unsigned long long)(uintptr_t)wlds;

        v4u g0;                                 // D# group 0 (128b)
        g0[0] = 1u;                             // count=1, is_restore=0, gather=0
        g0[1] = la;                             // lds_addr
        g0[2] = (unsigned int)(ga & 0xffffffffu);
        g0[3] = (unsigned int)((ga >> 32) & 0x1ffffffu) | (2u << 30); // type=2

        v8i g1;                                 // D# group 1 (256b)
        g1[0] = (2 << 16);                      // data_size=2 (4B), mask=0
        g1[1] = (int)((D4 & 0xffff) << 16);     // tensor_dim0[15:0]=512
        g1[2] = (int)(((D4 >> 16) & 0xffff) | ((36 & 0xffff) << 16)); // dim0 hi | tensor_dim1=36
        g1[3] = (int)(((36 >> 16) & 0xffff) | ((D4 & 0xffff) << 16)); // dim1 hi | tile_dim0=512
        g1[4] = 36;                             // tile_dim1=36, tile_dim2=0
        g1[5] = D4;                             // tensor_dim0_stride lo = 512
        g1[6] = (int)(((36 * D4) & 0xffff) << 16); // dim0_stride hi=0 | dim1_stride lo
        g1[7] = (int)((36 * D4) >> 16);         // dim1_stride hi

        v4i g2 = {0, 0, 0, 0};                  // unused (2D tile)
        v4i g3 = {0, 0, 0, 0};

        asm volatile("tensor_load_to_lds %0, %1, %2, %3"
                     :: "s"(g0), "s"(g1), "s"(g2), "s"(g3)
                     : "memory");
        __builtin_amdgcn_s_wait_tensorcnt(0);
    }
    __syncthreads();

    const int half = lane >> 4;
    const int mr   = lane & 15;
    const int p0   = (blockIdx.x * 8 + wv) * 16;
    const float* arow = u + (size_t)(p0 + mr) * D4;

    v8f acc[3] = {};
    for (int k0 = 0; k0 < D4; k0 += 32) {
        v16bf a;
#pragma unroll
        for (int e = 0; e < 16; ++e)
            a[e] = (__bf16)arow[k0 + kmap(e, half)];
#pragma unroll
        for (int nt = 0; nt < 3; ++nt) {
            const int n = nt * 16 + mr;
            v16bf bf;
#pragma unroll
            for (int e = 0; e < 16; ++e)
                bf[e] = (n < 36) ? (__bf16)wlds[n * D4 + k0 + kmap(e, half)]
                                 : (__bf16)0.0f;
            acc[nt] = __builtin_amdgcn_wmma_f32_16x16x32_bf16(
                false, a, false, bf, (short)0, acc[nt], false, false);
        }
    }
#pragma unroll
    for (int nt = 0; nt < 3; ++nt) {
        const int n = nt * 16 + mr;
#pragma unroll
        for (int r = 0; r < 8; ++r) {
            const int m = r + 8 * half;
            xdbl[(size_t)(p0 + m) * XDW + n] = acc[nt][r];
        }
    }
}

// ---------------------------------------------------------------------------
// Chunked parallel scan.  Recurrence h <- dA*h + w is linear, so:
//   pass A  : per (b,d4,n,chunk) compute P = prod(dA) and E = h_end from h=0
//   combine : per (b,d4,n) prefix  h <- P_c*h + E_c ; E_c <- h_start(c)
//   pass B  : per (b,d4,n,chunk) re-run chunk from h_start, emit y
// tid layout: n = tid&15, chunk = (tid>>4)&63, d = (tid>>10)&511, b = tid>>19
// => tid itself indexes the P/E buffers.
// ---------------------------------------------------------------------------
__global__ void k_scan_part(const float* __restrict__ xdbl, const float* __restrict__ u,
                            const float* __restrict__ dpw, const float* __restrict__ dpb,
                            const float* __restrict__ A_log,
                            float* __restrict__ P, float* __restrict__ E)
{
    const int tid = blockIdx.x * blockDim.x + threadIdx.x;
    if (tid >= BSZ * D4 * NCH * NST) return;
    const int n = tid & 15;
    const int c = (tid >> 4) & (NCH - 1);
    const int d = (tid >> 10) & (D4 - 1);
    const int b = tid >> 19;

    const float An = -__expf(A_log[d * NST + n]);
    const float w0 = dpw[d * 4 + 0], w1 = dpw[d * 4 + 1];
    const float w2 = dpw[d * 4 + 2], w3 = dpw[d * 4 + 3];
    const float db = dpb[d];

    const float* xd = xdbl + ((size_t)b * LL + (size_t)c * CHS) * XDW;
    const float* ub = u    + ((size_t)b * LL + (size_t)c * CHS) * D4 + d;

    float h = 0.0f, p = 1.0f;
    for (int s = 0; s < CHS; ++s) {
        const float* row = xd + (size_t)s * XDW;
        const float raw   = row[0] * w0 + row[1] * w1 + row[2] * w2 + row[3] * w3 + db;
        const float delta = softplus_f(raw);
        const float uv    = ub[(size_t)s * D4];
        const float Bt    = row[4 + n];
        const float dA    = __expf(delta * An);
        h = dA * h + (delta * uv) * Bt;
        p *= dA;
    }
    P[tid] = p;
    E[tid] = h;
}

__global__ void k_scan_combine(const float* __restrict__ P, float* __restrict__ E)
{
    const int tid = blockIdx.x * blockDim.x + threadIdx.x;
    if (tid >= BSZ * D4 * NST) return;
    const int n = tid & 15;
    const int d = (tid >> 4) & (D4 - 1);
    const int b = tid >> 13;

    const size_t base = ((size_t)(b * D4 + d) * NCH) * NST + n;
    float h = 0.0f;
    for (int c = 0; c < NCH; ++c) {
        const size_t i = base + (size_t)c * NST;
        const float e = E[i];
        const float p = P[i];
        E[i] = h;                    // corrected start state for chunk c
        h = p * h + e;
    }
}

__global__ void k_scan_apply(const float* __restrict__ xdbl, const float* __restrict__ dpw,
                             const float* __restrict__ dpb, const float* __restrict__ A_log,
                             const float* __restrict__ Dp, const float* __restrict__ E,
                             float* __restrict__ u)
{
    const int tid = blockIdx.x * blockDim.x + threadIdx.x;
    if (tid >= BSZ * D4 * NCH * NST) return;
    const int n = tid & 15;
    const int c = (tid >> 4) & (NCH - 1);
    const int d = (tid >> 10) & (D4 - 1);
    const int b = tid >> 19;

    const float An = -__expf(A_log[d * NST + n]);
    const float w0 = dpw[d * 4 + 0], w1 = dpw[d * 4 + 1];
    const float w2 = dpw[d * 4 + 2], w3 = dpw[d * 4 + 3];
    const float db = dpb[d];
    const float Dv = Dp[d];

    const float* xd = xdbl + ((size_t)b * LL + (size_t)c * CHS) * XDW;
    float* ub = u + ((size_t)b * LL + (size_t)c * CHS) * D4 + d;

    float h = E[tid];
    for (int s = 0; s < CHS; ++s) {
        const float* row = xd + (size_t)s * XDW;
        if (s + 4 < CHS) __builtin_prefetch(row + 4 * XDW, 0, 1);
        const float raw   = row[0] * w0 + row[1] * w1 + row[2] * w2 + row[3] * w3 + db;
        const float delta = softplus_f(raw);
        const float uv    = ub[(size_t)s * D4];
        const float Bt    = row[4 + n];
        const float Ct    = row[20 + n];
        const float dA    = __expf(delta * An);
        h = dA * h + (delta * uv) * Bt;
        float y = h * Ct;
        y += __shfl_xor(y, 1, 32);
        y += __shfl_xor(y, 2, 32);
        y += __shfl_xor(y, 4, 32);
        y += __shfl_xor(y, 8, 32);
        if (n == 0) ub[(size_t)s * D4] = y + uv * Dv;
    }
}

// ---------------------------------------------------------------------------
// Kernel 5: merge 4 directions (inverse perms) -> LayerNorm over D=128 ->
// * SiLU(z) -> WMMA out_proj [16 x 128] x [128 x 64] -> out (B, C_IN, H, W).
// ---------------------------------------------------------------------------
__global__ __launch_bounds__(32)
void k_out(const float* __restrict__ ys, const float* __restrict__ xz,
           const float* __restrict__ lng, const float* __restrict__ lnb,
           const float* __restrict__ opw, float* __restrict__ out)
{
    __shared__ float rowbuf[16][DD];
    __shared__ float mu[16], rs[16];

    const int lane = threadIdx.x;
    const int p0   = blockIdx.x * 16;

    for (int t = lane; t < 16 * DD; t += 32) {
        const int r = t >> 7, d = t & (DD - 1);
        const int p = p0 + r, b = p >> 12, l = p & (LL - 1);
        const int tl = ((l & 63) << 6) | (l >> 6);
        const float* yb = ys + (size_t)b * LL * D4;
        rowbuf[r][d] = yb[(size_t)l           * D4 +          d]
                     + yb[(size_t)tl          * D4 + DD     + d]
                     + yb[(size_t)(4095 - l)  * D4 + 2 * DD + d]
                     + yb[(size_t)(4095 - tl) * D4 + 3 * DD + d];
    }
    __syncthreads();

    if (lane < 16) {
        float s = 0.0f, s2 = 0.0f;
        for (int d = 0; d < DD; ++d) {
            const float v = rowbuf[lane][d];
            s += v; s2 += v * v;
        }
        const float m   = s * (1.0f / DD);
        const float var = s2 * (1.0f / DD) - m * m;
        mu[lane] = m;
        rs[lane] = rsqrtf(var + 1e-5f);
    }
    __syncthreads();

    for (int t = lane; t < 16 * DD; t += 32) {
        const int r = t >> 7, d = t & (DD - 1);
        const int p = p0 + r;
        const float zv = xz[(size_t)p * E2D + DD + d];
        const float v  = (rowbuf[r][d] - mu[r]) * rs[r] * lng[d] + lnb[d];
        rowbuf[r][d] = v * silu_f(zv);
    }
    __syncthreads();

    const int half = lane >> 4;
    const int mr   = lane & 15;
    v8f acc[4] = {};
#pragma unroll
    for (int k0 = 0; k0 < DD; k0 += 32) {
        v16bf a;
#pragma unroll
        for (int e = 0; e < 16; ++e)
            a[e] = (__bf16)rowbuf[mr][k0 + kmap(e, half)];
#pragma unroll
        for (int nt = 0; nt < 4; ++nt) {
            const int n = nt * 16 + mr;
            v16bf bf;
#pragma unroll
            for (int e = 0; e < 16; ++e)
                bf[e] = (__bf16)opw[n * DD + k0 + kmap(e, half)];
            acc[nt] = __builtin_amdgcn_wmma_f32_16x16x32_bf16(
                false, a, false, bf, (short)0, acc[nt], false, false);
        }
    }

#pragma unroll
    for (int nt = 0; nt < 4; ++nt) {
        const int c = nt * 16 + mr;
#pragma unroll
        for (int r = 0; r < 8; ++r) {
            const int m = r + 8 * half;
            const int p = p0 + m, b = p >> 12, l = p & (LL - 1);
            out[(size_t)(b * CIN + c) * LL + l] = acc[nt][r];
        }
    }
}

// ---------------------------------------------------------------------------
extern "C" void kernel_launch(void* const* d_in, const int* in_sizes, int n_in,
                              void* d_out, int out_size, void* d_ws, size_t ws_size,
                              hipStream_t stream)
{
    const float* x    = (const float*)d_in[0];
    const float* ipw  = (const float*)d_in[1];
    const float* ipb  = (const float*)d_in[2];
    const float* cw   = (const float*)d_in[3];
    const float* cb   = (const float*)d_in[4];
    const float* xpw  = (const float*)d_in[5];
    const float* dpw  = (const float*)d_in[6];
    const float* dpb  = (const float*)d_in[7];
    const float* Alog = (const float*)d_in[8];
    const float* Dp   = (const float*)d_in[9];
    const float* lng  = (const float*)d_in[10];
    const float* lnb  = (const float*)d_in[11];
    const float* opw  = (const float*)d_in[12];
    float* out = (float*)d_out;

    // ws (floats): xz [16384*256] | u/ys [16384*512] | xdbl [16384*48]
    //            | P [2M] | E [2M]   (~70 MB total)
    float* ws   = (float*)d_ws;
    float* xz   = ws;
    float* u    = xz   + (size_t)MROWS * E2D;
    float* xdbl = u    + (size_t)MROWS * D4;
    float* Pb   = xdbl + (size_t)MROWS * XDW;
    float* Eb   = Pb   + (size_t)BSZ * D4 * NCH * NST;

    const int scanN = BSZ * D4 * NCH * NST;     // 2,097,152

    k_inproj<<<dim3(MROWS / 16, 4), 32, 0, stream>>>(x, ipw, ipb, xz);
    k_conv_u<<<(BSZ * DD * LL + 255) / 256, 256, 0, stream>>>(xz, cw, cb, u);
    k_xdbl<<<MROWS / (16 * 8), 256, 0, stream>>>(u, xpw, xdbl);
    k_scan_part<<<scanN / 256, 256, 0, stream>>>(xdbl, u, dpw, dpb, Alog, Pb, Eb);
    k_scan_combine<<<(BSZ * D4 * NST) / 256, 256, 0, stream>>>(Pb, Eb);
    k_scan_apply<<<scanN / 256, 256, 0, stream>>>(xdbl, dpw, dpb, Alog, Dp, Eb, u);
    k_out<<<MROWS / 16, 32, 0, stream>>>(u, xz, lng, lnb, opw, out);
}
